// LSTM_51445118271782
// MI455X (gfx1250) — matrix operand order repaired and checked by reading, not compiled, not run on hardware
//
#include <hip/hip_runtime.h>

#define DEVINL __device__ __forceinline__

typedef __attribute__((ext_vector_type(16))) __bf16          v16bf;
typedef __attribute__((ext_vector_type(8)))  float           v8f;
typedef __attribute__((ext_vector_type(8)))  unsigned        v8u;
typedef int gv4i __attribute__((vector_size(16)));           // matches builtin's V4i

constexpr int B_   = 64;        // batch
constexpr int S_   = 1024;      // seq
constexpr int K_   = 512;       // input size == hidden size
constexpr int H_   = 512;       // hidden
constexpr int GN_  = 2048;      // 4*H gate columns
constexpr int M_   = B_ * S_;   // 65536 rows of x
constexpr int KCH  = K_ / 32;   // 16 K-chunks of 32
constexpr int NT_  = GN_ / 16;  // 128 n-tiles of 16
constexpr int NBLK2 = 16;       // persistent blocks in recurrent phase
constexpr int WELEM = K_ * GN_;     // 1<<20 elements per weight matrix
constexpr int WHALF = WELEM / 2;    // dwords per hi (or lo) packed region

// ---------- CDNA5 async global->LDS (ASYNCcnt) with synchronous fallback ----------
#if defined(__HIP_DEVICE_COMPILE__) && __has_builtin(__builtin_amdgcn_global_load_async_to_lds_b128)
#define HAVE_ASYNC_LDS 1
#else
#define HAVE_ASYNC_LDS 0
#endif

#if HAVE_ASYNC_LDS
DEVINL void async_copy16(const void* g, void* l) {
    __builtin_amdgcn_global_load_async_to_lds_b128(
        (__attribute__((address_space(1))) gv4i*)(void*)g,
        (__attribute__((address_space(3))) gv4i*)l, 0, 0);
}
DEVINL void wait_async0() {
#if __has_builtin(__builtin_amdgcn_s_wait_asynccnt)
    __builtin_amdgcn_s_wait_asynccnt(0);
#else
    asm volatile("s_wait_asynccnt 0" ::: "memory");
#endif
}
#else
DEVINL void wait_async0() {}
#endif

// ---------- bf16 split helpers (bf16x3 error-corrected fp32 emulation) ----------
DEVINL unsigned short f2bf_rne(float f) {
    unsigned u = __builtin_bit_cast(unsigned, f);
    unsigned r = u + 0x7FFFu + ((u >> 16) & 1u);      // round-to-nearest-even
    return (unsigned short)(r >> 16);
}
DEVINL float bf2f(unsigned short h) {
    unsigned u = ((unsigned)h) << 16;
    return __builtin_bit_cast(float, u);
}
// Round two fp32 to bf16 hi + bf16 lo; pack pairs into dwords (elem0 low half).
DEVINL void splitpair(float f0, float f1, unsigned& hi, unsigned& lo) {
    unsigned u0 = __builtin_bit_cast(unsigned, f0);
    unsigned r0 = u0 + 0x7FFFu + ((u0 >> 16) & 1u);
    unsigned u1 = __builtin_bit_cast(unsigned, f1);
    unsigned r1 = u1 + 0x7FFFu + ((u1 >> 16) & 1u);
    hi = (r0 >> 16) | (r1 & 0xFFFF0000u);
    float h0 = __builtin_bit_cast(float, r0 & 0xFFFF0000u);
    float h1 = __builtin_bit_cast(float, r1 & 0xFFFF0000u);
    float l0 = f0 - h0;
    float l1 = f1 - h1;
    unsigned v0 = __builtin_bit_cast(unsigned, l0);
    unsigned s0 = v0 + 0x7FFFu + ((v0 >> 16) & 1u);
    unsigned v1 = __builtin_bit_cast(unsigned, l1);
    unsigned s1 = v1 + 0x7FFFu + ((v1 >> 16) & 1u);
    lo = (s0 >> 16) | (s1 & 0xFFFF0000u);
}

DEVINL v8f wmma_bf16(v16bf a, v16bf b, v8f c) {
    // D = A(16x32 bf16) * B(32x16 bf16) + C(16x16 f32)
    return __builtin_amdgcn_wmma_f32_16x16x32_bf16(false, a, false, b, (short)0, c, false, false);
}

// Build A hi/lo fragments for one 16x32 tile from fp32 row-major source.
// ISA 7.12.2 16-bit A layout: lanes 0-15 row M=lane, K = kc+{0..7,16..23};
// lanes 16-31 row M=lane-16, K = kc+{8..15,24..31}.
DEVINL void cvt_a(const float* __restrict__ row, int kb, int koff,
                  v16bf& ah, v16bf& al) {
    float af[16] __attribute__((aligned(16)));
    *(float4*)(af + 0)  = *(const float4*)(row + kb + koff);
    *(float4*)(af + 4)  = *(const float4*)(row + kb + koff + 4);
    *(float4*)(af + 8)  = *(const float4*)(row + kb + 16 + koff);
    *(float4*)(af + 12) = *(const float4*)(row + kb + 16 + koff + 4);
    v8u hu, lu;
#pragma unroll
    for (int d = 0; d < 8; ++d) {
        unsigned h, l;
        splitpair(af[2 * d], af[2 * d + 1], h, l);
        hu[d] = h;
        lu[d] = l;
    }
    ah = __builtin_bit_cast(v16bf, hu);
    al = __builtin_bit_cast(v16bf, lu);
}

DEVINL float sigmoidf_fast(float x) { return 1.0f / (1.0f + __expf(-x)); }
DEVINL float tanhf_fast(float x) {
    x = fminf(fmaxf(x, -15.0f), 15.0f);
    float e = __expf(2.0f * x);
    return (e - 1.0f) / (e + 1.0f);
}

// ---------- Phase 0: pack W_i / W_h into B-fragment-ordered bf16 hi/lo regions.
// Region layout (dwords): hi frags at [0, WHALF), lo frags at [WHALF, WELEM).
// Frag f = kcIdx*128 + nTile occupies 256 dwords: lane*8 + d, dword d = elements
// {2d, 2d+1}, element e -> (k = 32*kcIdx + 8*(lane>=16) + e + 8*(e>=8), n = 16*nTile + lane&15).
__global__ __launch_bounds__(256) void pack_weights(
    const float* __restrict__ Wi, const float* __restrict__ Wh,
    unsigned* __restrict__ Pi, unsigned* __restrict__ Ph) {
    int tid = blockIdx.x * 256 + threadIdx.x;        // 2 * WHALF threads
    const float* W = Wi;
    unsigned* P = Pi;
    int idx = tid;
    if (tid >= WHALF) { W = Wh; P = Ph; idx -= WHALF; }
    int d     = idx & 7;
    int lane  = (idx >> 3) & 31;
    int f     = idx >> 8;                            // kcIdx*128 + nTile
    int nTile = f & 127;
    int kcIdx = f >> 7;
    int kbase = kcIdx * 32 + ((lane >> 4) << 3) + ((d >> 2) << 3);
    int n     = nTile * 16 + (lane & 15);
    float f0 = W[(kbase + 2 * d)     * GN_ + n];
    float f1 = W[(kbase + 2 * d + 1) * GN_ + n];
    unsigned hi, lo;
    splitpair(f0, f1, hi, lo);
    P[idx]         = hi;
    P[WHALF + idx] = lo;
}

// stage one K-chunk of 4 B-frags (hi+lo, 8 KB) into an LDS buffer
DEVINL void stage_chunk(const unsigned* __restrict__ Pi, int kc, int nTile0,
                        unsigned* dstLds, int tid) {
    const uint4* srcH = (const uint4*)(Pi + ((long)kc * NT_ + nTile0) * 256);
    const uint4* srcL = (const uint4*)(Pi + WHALF + ((long)kc * NT_ + nTile0) * 256);
    uint4* dH = (uint4*)dstLds + tid;
    uint4* dL = (uint4*)(dstLds + 1024) + tid;
#if HAVE_ASYNC_LDS
    async_copy16(srcH + tid, dH);
    async_copy16(srcL + tid, dL);
#else
    *dH = srcH[tid];
    *dL = srcL[tid];
#endif
}

// ---------- Phase 1: x_proj[m, 0:2048] = x[m, 0:512] @ W_i + b ----------
// Block = 8 waves, tile 128M x 64N. grid = (32 N-strips fastest, 512 M-blocks)
// so x rows stay L2-hot across all N-strips. B-frag staging is double-buffered
// async-to-LDS, prefetching chunk kc+1 while WMMAs consume chunk kc.
__global__ __launch_bounds__(256) void input_proj(
    const float* __restrict__ x, const unsigned* __restrict__ Pi,
    const float* __restrict__ bias, float* __restrict__ xproj) {
    __shared__ unsigned sB[2][2048];                 // ping-pong: hi 1024 dw + lo 1024 dw
    const int nStrip = blockIdx.x;                   // 0..31
    const int mBlock = blockIdx.y;                   // 0..511
    const int wave = threadIdx.x >> 5;
    const int lane = threadIdx.x & 31;
    const int rowBase = mBlock * 128 + wave * 16;
    const int nTile0 = nStrip * 4;

    v8f acc[4] = {};
    const float* xr = x + (long)(rowBase + (lane & 15)) * K_;
    const int koff = (lane >> 4) << 3;

    stage_chunk(Pi, 0, nTile0, sB[0], threadIdx.x);
    for (int kc = 0; kc < KCH; ++kc) {
        unsigned* cur = sB[kc & 1];
        wait_async0();
        __syncthreads();                             // staged data visible; prev compute done
        if (kc + 1 < KCH) stage_chunk(Pi, kc + 1, nTile0, sB[(kc + 1) & 1], threadIdx.x);

        v16bf ah, al;
        cvt_a(xr, kc * 32, koff, ah, al);
#pragma unroll
        for (int nt = 0; nt < 4; ++nt) {
            v16bf bh = __builtin_bit_cast(v16bf, *(const v8u*)(cur + nt * 256 + lane * 8));
            v16bf bl = __builtin_bit_cast(v16bf, *(const v8u*)(cur + 1024 + nt * 256 + lane * 8));
            acc[nt] = wmma_bf16(ah, bh, acc[nt]);
            acc[nt] = wmma_bf16(al, bh, acc[nt]);
            acc[nt] = wmma_bf16(ah, bl, acc[nt]);
        }
    }
    // C/D layout: acc[nt][r] = C[row = r + 8*(lane>=16), col = lane&15]
    const int rowOff = rowBase + ((lane >> 4) << 3);
    const int col0 = nStrip * 64 + (lane & 15);
#pragma unroll
    for (int nt = 0; nt < 4; ++nt) {
        int col = col0 + nt * 16;
        float bv = bias[col];
#pragma unroll
        for (int r = 0; r < 8; ++r)
            xproj[(long)(rowOff + r) * GN_ + col] = acc[nt][r] + bv;
    }
}

// ---------- Phase 2: persistent recurrent scan ----------
// 16 blocks x 256 threads; block owns 32 hidden cols; its 256 KB W_h slice lives
// in LDS (CDNA5: 320 KB/WGP). h is double-buffered in global as SPLIT bf16
// (hi[64*512], lo[64*512] per buffer), stored pre-swizzled into A-fragment
// element order so each step's A load is one 32B vector load per K-chunk.
// x_proj gathers for step t+1 are issued BEFORE the grid barrier of step t so
// their memory latency overlaps the barrier wait.
__global__ __launch_bounds__(256) void lstm_scan(
    const float* __restrict__ xproj, const unsigned* __restrict__ Ph,
    unsigned short* __restrict__ hbuf, unsigned* __restrict__ bar,
    float* __restrict__ out) {
    extern __shared__ unsigned sW[];                 // 128 slots * 512 dw = 256 KB
    const int blk  = blockIdx.x;
    const int wave = threadIdx.x >> 5;
    const int lane = threadIdx.x & 31;
    const int m = wave & 3;
    const int g = wave >> 2;

    // stage W_h slice once: slot s=(grp*4+q)*16+kc holds hi(256dw) then lo(256dw)
    for (int u4 = threadIdx.x; u4 < 16384; u4 += 256) {
        int dw     = u4 * 4;
        int slot   = dw >> 9;
        int inslot = dw & 511;
        int half   = inslot >> 8;                    // 0 = hi, 1 = lo
        int off    = inslot & 255;
        int kc  = slot & 15;
        int q   = (slot >> 4) & 3;
        int grp = slot >> 6;
        int f = kc * NT_ + (q * 32 + blk * 2 + grp);
        const uint4* src = (const uint4*)(Ph + (long)half * WHALF + (long)f * 256 + off);
        uint4* dst = (uint4*)sW + u4;
#if HAVE_ASYNC_LDS
        async_copy16(src, dst);
#else
        *dst = *src;
#endif
    }
    wait_async0();
    __syncthreads();

    v8f cfrag = {}, hfrag = {};
    const int arowb = m * 16 + (lane & 15);          // batch row this lane loads
    const int aoff  = (lane >> 4) << 4;              // swizzled element offset in chunk
    const int crow  = m * 16 + ((lane >> 4) << 3);   // C-layout batch row base
    const int ccol  = blk * 32 + g * 16 + (lane & 15);
    // swizzled store position for hidden col ccol (A-frag element order)
    const int cc = ccol & 31;
    const int hswz = (ccol >> 5) * 32 + (((cc >> 3) & 1) << 4) + ((cc & 7) | ((cc >> 4) << 3));

    unsigned* cnt = bar;
    unsigned* gen = bar + 1;

    float xpc[4][8], xpn[4][8];
#pragma unroll
    for (int q = 0; q < 4; ++q)
#pragma unroll
        for (int r = 0; r < 8; ++r)
            xpc[q][r] = xproj[((long)(crow + r) * S_ + 0) * GN_ + q * H_ + ccol];

    for (int t = 0; t < S_; ++t) {
        const unsigned short* hpH = hbuf + (long)(t & 1) * (2 * B_ * H_);
        const unsigned short* hpL = hpH + B_ * H_;
        unsigned short* hnH = hbuf + (long)((t + 1) & 1) * (2 * B_ * H_);
        unsigned short* hnL = hnH + B_ * H_;

        v8f acc[4] = {};
        const unsigned short* hrH = hpH + arowb * H_;
        const unsigned short* hrL = hpL + arowb * H_;
        for (int kc = 0; kc < KCH; ++kc) {
            v16bf ah = __builtin_bit_cast(v16bf, *(const v8u*)(hrH + kc * 32 + aoff));
            v16bf al = __builtin_bit_cast(v16bf, *(const v8u*)(hrL + kc * 32 + aoff));
#pragma unroll
            for (int q = 0; q < 4; ++q) {
                const unsigned* slotp = sW + ((g * 4 + q) * 16 + kc) * 512 + lane * 8;
                v16bf bh = __builtin_bit_cast(v16bf, *(const v8u*)(slotp));
                v16bf bl = __builtin_bit_cast(v16bf, *(const v8u*)(slotp + 256));
                acc[q] = wmma_bf16(ah, bh, acc[q]);
                acc[q] = wmma_bf16(al, bh, acc[q]);
                acc[q] = wmma_bf16(ah, bl, acc[q]);
            }
        }
        // elementwise LSTM cell on this wave's 16x16 tile
#pragma unroll
        for (int r = 0; r < 8; ++r) {
            float gi = acc[0][r] + xpc[0][r];
            float gf = acc[1][r] + xpc[1][r];
            float gg = acc[2][r] + xpc[2][r];
            float go = acc[3][r] + xpc[3][r];
            float it = sigmoidf_fast(gi);
            float ft = sigmoidf_fast(gf);
            float gt = tanhf_fast(gg);
            float ot = sigmoidf_fast(go);
            float c  = ft * cfrag[r] + it * gt;
            float h  = ot * tanhf_fast(c);
            cfrag[r] = c;
            hfrag[r] = h;
            unsigned short hh = f2bf_rne(h);
            unsigned short hl = f2bf_rne(h - bf2f(hh));
            hnH[(crow + r) * H_ + hswz] = hh;
            hnL[(crow + r) * H_ + hswz] = hl;
        }
        // prefetch x_proj for step t+1; latency overlaps the grid barrier below
        if (t + 1 < S_) {
#pragma unroll
            for (int q = 0; q < 4; ++q)
#pragma unroll
                for (int r = 0; r < 8; ++r)
                    xpn[q][r] = xproj[((long)(crow + r) * S_ + (t + 1)) * GN_ + q * H_ + ccol];
        }
        // grid barrier (sense via monotonically increasing generation)
        __threadfence();
        __syncthreads();
        if (threadIdx.x == 0) {
            unsigned old = __hip_atomic_load(gen, __ATOMIC_RELAXED, __HIP_MEMORY_SCOPE_AGENT);
            unsigned arr = __hip_atomic_fetch_add(cnt, 1u, __ATOMIC_ACQ_REL, __HIP_MEMORY_SCOPE_AGENT);
            if (arr == NBLK2 - 1) {
                __hip_atomic_store(cnt, 0u, __ATOMIC_RELAXED, __HIP_MEMORY_SCOPE_AGENT);
                __hip_atomic_fetch_add(gen, 1u, __ATOMIC_RELEASE, __HIP_MEMORY_SCOPE_AGENT);
            } else {
                while (__hip_atomic_load(gen, __ATOMIC_ACQUIRE, __HIP_MEMORY_SCOPE_AGENT) == old)
                    __builtin_amdgcn_s_sleep(1);
            }
        }
        __syncthreads();
#pragma unroll
        for (int q = 0; q < 4; ++q)
#pragma unroll
            for (int r = 0; r < 8; ++r)
                xpc[q][r] = xpn[q][r];
    }
    // outputs: h then c, each [64, 512] fp32
#pragma unroll
    for (int r = 0; r < 8; ++r) {
        out[(crow + r) * H_ + ccol] = hfrag[r];
        out[B_ * H_ + (crow + r) * H_ + ccol] = cfrag[r];
    }
}

extern "C" void kernel_launch(void* const* d_in, const int* in_sizes, int n_in,
                              void* d_out, int out_size, void* d_ws, size_t ws_size,
                              hipStream_t stream) {
    (void)in_sizes; (void)n_in; (void)out_size; (void)ws_size;
    const float* x  = (const float*)d_in[0];
    const float* Wi = (const float*)d_in[1];
    const float* Wh = (const float*)d_in[2];
    const float* b  = (const float*)d_in[3];

    char* ws = (char*)d_ws;
    unsigned*       bar   = (unsigned*)ws;                       // 256 B
    unsigned short* hbuf  = (unsigned short*)(ws + 256);         // 2 bufs * (hi+lo) = 256 KB
    unsigned*       Pi    = (unsigned*)(ws + 256 + 262144);      // 4 MB
    unsigned*       Ph    = Pi + WELEM;                          // 4 MB
    float*          xproj = (float*)((char*)(Ph + WELEM));       // 512 MB

    // zero barrier + both h buffers every launch (deterministic)
    (void)hipMemsetAsync(ws, 0, 256 + 262144, stream);

    pack_weights<<<(2 * WHALF) / 256, 256, 0, stream>>>(Wi, Wh, Pi, Ph);

    dim3 g1(32, M_ / 128);                                       // N-strips fastest
    input_proj<<<g1, 256, 0, stream>>>(x, Pi, b, xproj);

    static bool attr_set = false;
    if (!attr_set) {
        (void)hipFuncSetAttribute((const void*)lstm_scan,
                                  hipFuncAttributeMaxDynamicSharedMemorySize, 262144);
        attr_set = true;
    }
    lstm_scan<<<NBLK2, 256, 262144, stream>>>(xproj, Ph, hbuf, bar, (float*)d_out);
}
